// MultiheadSelfAttention_1288490189606
// MI455X (gfx1250) — matrix operand order repaired and checked by reading, not compiled, not run on hardware
//
#include <hip/hip_runtime.h>

typedef __bf16 bf16;
typedef __attribute__((ext_vector_type(8)))  __bf16 bf16x8;
typedef __attribute__((ext_vector_type(16))) __bf16 bf16x16;
typedef __attribute__((ext_vector_type(8)))  float  floatx8;
typedef __attribute__((ext_vector_type(4)))  int    v4i;

#define DMODEL 768
#define NHEAD  12
#define DK     64
#define SEQ    4096

// ---- CDNA5 async global->LDS path (guarded; sync fallback) ---------------
#if __has_builtin(__builtin_amdgcn_global_load_async_to_lds_b128) && \
    __has_builtin(__builtin_amdgcn_s_wait_asynccnt)
#define HAVE_ASYNC_LDS 1
typedef __attribute__((address_space(1))) v4i gv4i;
typedef __attribute__((address_space(3))) v4i lv4i;
#endif

__device__ __forceinline__ void cp16_to_lds(const bf16* g, bf16* l) {
#ifdef HAVE_ASYNC_LDS
  __builtin_amdgcn_global_load_async_to_lds_b128((gv4i*)g, (lv4i*)l, 0, 0);
#else
  *(bf16x8*)l = *(const bf16x8*)g;
#endif
}
__device__ __forceinline__ void async_wait() {
#ifdef HAVE_ASYNC_LDS
  __builtin_amdgcn_s_wait_asynccnt(0);
#endif
}

// ---- helpers -------------------------------------------------------------

// Build a 16-element bf16 operand vector from two contiguous 8-element chunks.
__device__ __forceinline__ bf16x16 ld16(const bf16* p0, const bf16* p1) {
  bf16x8 lo = *(const bf16x8*)p0;
  bf16x8 hi = *(const bf16x8*)p1;
  bf16x16 r;
#pragma unroll
  for (int i = 0; i < 8; ++i) { r[i] = lo[i]; r[i + 8] = hi[i]; }
  return r;
}

__device__ __forceinline__ floatx8 wmma_bf16(bf16x16 a, bf16x16 b, floatx8 c) {
  return __builtin_amdgcn_wmma_f32_16x16x32_bf16(false, a, false, b,
                                                 (short)0, c, false, false);
}

__device__ __forceinline__ float redmax16(float v) {
#pragma unroll
  for (int m = 8; m >= 1; m >>= 1) v = fmaxf(v, __shfl_xor(v, m, 32));
  return v;
}
__device__ __forceinline__ float redsum16(float v) {
#pragma unroll
  for (int m = 8; m >= 1; m >>= 1) v += __shfl_xor(v, m, 32);
  return v;
}

// ---- kernel 0: fp32 -> bf16 conversion -----------------------------------

__global__ void cvt_bf16(const float* __restrict__ src, bf16* __restrict__ dst,
                         int n) {
  int i = blockIdx.x * blockDim.x + threadIdx.x;
  if (i < n) dst[i] = (bf16)src[i];
}

// ---- kernel 1: QKV projection + RoPE -------------------------------------
// grid: 3 mats * 256 row-tiles * 3 col-groups; 128 threads (4 waves).
// Each wave owns a 16(seq) x 64(out) tile == one full head slice:
// 4 f32 accumulators, one A-operand load amortized over 4 WMMAs per k-step.
__global__ void qkv_rope_kernel(const bf16* __restrict__ xb,
                                const bf16* __restrict__ Wqb,
                                const bf16* __restrict__ Wkb,
                                const bf16* __restrict__ Wvb,
                                const int*  __restrict__ tpos,
                                bf16* __restrict__ Qb,   // [12][4096][64]
                                bf16* __restrict__ Kb,   // [12][4096][64]
                                bf16* __restrict__ Vtb)  // [12][64][4096]
{
  const int lane = threadIdx.x & 31;
  const int wid  = threadIdx.x >> 5;
  const int hl   = lane >> 4;
  const int ln   = lane & 15;

  int bx = blockIdx.x;
  const int mat     = bx / (256 * 3);
  int rem           = bx % (256 * 3);
  const int rowtile = rem / 3;
  const int h       = (rem % 3) * 4 + wid;   // head index == 64-col chunk
  const int s0 = rowtile * 16;
  const int o0 = h * DK;

  const bf16* W = (mat == 0) ? Wqb : (mat == 1) ? Wkb : Wvb;

  floatx8 acc[4];
#pragma unroll
  for (int t = 0; t < 4; ++t)
    acc[t] = (floatx8){0.f,0.f,0.f,0.f,0.f,0.f,0.f,0.f};

  const bf16* arow = xb + (size_t)(s0 + ln) * DMODEL;
  const bf16* brow[4];
#pragma unroll
  for (int t = 0; t < 4; ++t)
    brow[t] = W + (size_t)(o0 + t * 16 + ln) * DMODEL;

#pragma unroll 2
  for (int kk = 0; kk < DMODEL; kk += 32) {
    bf16x16 a = ld16(arow + kk + hl * 8, arow + kk + 16 + hl * 8);
#pragma unroll
    for (int t = 0; t < 4; ++t) {
      bf16x16 b = ld16(brow[t] + kk + hl * 8, brow[t] + kk + 16 + hl * 8);
      acc[t] = wmma_bf16(a, b, acc[t]);
    }
  }

  if (mat < 2) {
    // RoPE in-register: pair partner (channel c^1) sits in lane^1.
#pragma unroll
    for (int t = 0; t < 4; ++t) {
      const int   c    = t * 16 + ln;
      const int   pair = c >> 1;
      const float invf =
          __expf(-9.2103403719761836f * (float)(2 * pair) / 64.f);
      const float sgn = (c & 1) ? 1.0f : -1.0f;
#pragma unroll
      for (int r = 0; r < 8; ++r) {
        float v = acc[t][r];
        float partner = __shfl_xor(v, 1, 32);
        int srow = s0 + r + 8 * hl;
        float ang = (float)tpos[srow] * invf;
        float sn, cs; __sincosf(ang, &sn, &cs);
        acc[t][r] = v * cs + partner * (sgn * sn);
      }
    }
  }
  if (mat == 0) {
#pragma unroll
    for (int t = 0; t < 4; ++t)
#pragma unroll
      for (int r = 0; r < 8; ++r) acc[t][r] *= 0.125f;  // 1/sqrt(64)
  }

  if (mat < 2) {
    bf16* dst = (mat == 0 ? Qb : Kb) + (size_t)h * SEQ * DK;
#pragma unroll
    for (int t = 0; t < 4; ++t)
#pragma unroll
      for (int r = 0; r < 8; ++r)
        dst[(size_t)(s0 + r + 8 * hl) * DK + t * 16 + ln] = (bf16)acc[t][r];
  } else {
#pragma unroll
    for (int t = 0; t < 4; ++t)
#pragma unroll
      for (int r = 0; r < 8; ++r)
        Vtb[((size_t)h * DK + t * 16 + ln) * SEQ + (s0 + r + 8 * hl)] =
            (bf16)acc[t][r];
  }
}

// ---- kernel 2: causal flash attention ------------------------------------
// grid: 12 heads * 64 query-blocks; 128 threads (4 waves), each wave:
// 16 queries x dk=64. K/V tiles for the 32-key step are staged once per
// block into LDS via async global->LDS loads and shared by all 4 waves.
__global__ void attention_kernel(const bf16* __restrict__ Qb,
                                 const bf16* __restrict__ Kb,
                                 const bf16* __restrict__ Vtb,
                                 bf16* __restrict__ attnb)  // [4096][768]
{
  __shared__ __align__(16) bf16 Kt[32][64];       // 32 keys x dk
  __shared__ __align__(16) bf16 Vt[64][32];       // dk x 32 keys
  __shared__ __align__(16) bf16 pbuf[4][16][32];  // per-wave P staging

  const int tid  = threadIdx.x;
  const int lane = tid & 31;
  const int wid  = tid >> 5;
  const int hl   = lane >> 4;
  const int ln   = lane & 15;

  const int h      = blockIdx.x / 64;
  const int qblock = blockIdx.x % 64;
  const int q0     = qblock * 64 + wid * 16;

  const bf16* Qh = Qb  + (size_t)h * SEQ * DK;
  const bf16* Kh = Kb  + (size_t)h * SEQ * DK;
  const bf16* Vh = Vtb + (size_t)h * DK * SEQ;

  // Preload the Q A-operand tiles (fixed across the key loop).
  const bf16* qrow = Qh + (size_t)(q0 + ln) * DK;
  bf16x16 qa0 = ld16(qrow + 0  + hl * 8, qrow + 16 + hl * 8);
  bf16x16 qa1 = ld16(qrow + 32 + hl * 8, qrow + 48 + hl * 8);

  floatx8 O[4];
#pragma unroll
  for (int d = 0; d < 4; ++d) O[d] = (floatx8){0.f,0.f,0.f,0.f,0.f,0.f,0.f,0.f};
  float m[8], lsum[8];
#pragma unroll
  for (int r = 0; r < 8; ++r) { m[r] = -1e30f; lsum[r] = 0.f; }

  const int nkb = (qblock + 1) * 2;  // key blocks of 32 (uniform over block)
  for (int kb = 0; kb < nkb; ++kb) {
    const int k0 = kb * 32;

    __syncthreads();  // previous iteration's LDS readers are done
    // Stage K tile (32x64 = 256 16B chunks) and Vt tile (64x32 = 256 chunks):
    // 2 chunks of each per thread, via async global->LDS.
    {
      const int c0 = tid * 2;
#pragma unroll
      for (int i = 0; i < 2; ++i) {
        const int c  = c0 + i;
        const int kr = c >> 3, ko = (c & 7) * 8;
        cp16_to_lds(Kh + (size_t)(k0 + kr) * DK + ko, &Kt[kr][ko]);
        const int vr = c >> 2, vo = (c & 3) * 8;
        cp16_to_lds(Vh + (size_t)vr * SEQ + k0 + vo, &Vt[vr][vo]);
      }
    }
    async_wait();
    __syncthreads();  // staged tiles visible block-wide

    // --- scores: two 16x16 tiles over 32 keys, contraction over dk=64
    floatx8 sA = {0.f,0.f,0.f,0.f,0.f,0.f,0.f,0.f};
    floatx8 sB = {0.f,0.f,0.f,0.f,0.f,0.f,0.f,0.f};
    {
      const bf16* kr = &Kt[ln][0];
      sA = wmma_bf16(qa0, ld16(kr + hl * 8,      kr + 16 + hl * 8),      sA);
      sA = wmma_bf16(qa1, ld16(kr + 32 + hl * 8, kr + 48 + hl * 8), sA);
    }
    {
      const bf16* kr = &Kt[16 + ln][0];
      sB = wmma_bf16(qa0, ld16(kr + hl * 8,      kr + 16 + hl * 8),      sB);
      sB = wmma_bf16(qa1, ld16(kr + 32 + hl * 8, kr + 48 + hl * 8), sB);
    }

    // --- online softmax update (per C-layout row)
#pragma unroll
    for (int r = 0; r < 8; ++r) {
      const int q    = q0 + r + 8 * hl;
      const int keyA = k0 + ln;
      const int keyB = k0 + 16 + ln;
      float sa = (keyA <= q) ? sA[r] : -1e30f;
      float sb = (keyB <= q) ? sB[r] : -1e30f;
      float mb   = redmax16(fmaxf(sa, sb));
      float mnew = fmaxf(m[r], mb);
      float alpha = __expf(m[r] - mnew);
      float pa = __expf(sa - mnew);
      float pb = __expf(sb - mnew);
      lsum[r] = lsum[r] * alpha + redsum16(pa + pb);
      m[r] = mnew;
#pragma unroll
      for (int d = 0; d < 4; ++d) O[d][r] *= alpha;
      pbuf[wid][r + 8 * hl][ln]      = (bf16)pa;
      pbuf[wid][r + 8 * hl][ln + 16] = (bf16)pb;
    }
    __syncthreads();

    // --- P re-read in A-operand layout; V B-operand straight from LDS
    const bf16* prow = &pbuf[wid][ln][0];
    bf16x16 pA = ld16(prow + hl * 8, prow + 16 + hl * 8);
#pragma unroll
    for (int d = 0; d < 4; ++d) {
      const bf16* vr = &Vt[d * 16 + ln][0];
      bf16x16 vb = ld16(vr + hl * 8, vr + 16 + hl * 8);
      O[d] = wmma_bf16(pA, vb, O[d]);
    }
  }

  // normalize + store [s][h*64 + d] bf16
#pragma unroll
  for (int d = 0; d < 4; ++d)
#pragma unroll
    for (int r = 0; r < 8; ++r)
      attnb[(size_t)(q0 + r + 8 * hl) * DMODEL + h * DK + d * 16 + ln] =
          (bf16)(O[d][r] / lsum[r]);
}

// ---- kernel 3: output projection -----------------------------------------
// grid: 256 row-tiles * 3 col-groups; 128 threads; 16x64 per wave.
__global__ void out_proj_kernel(const bf16* __restrict__ attnb,
                                const bf16* __restrict__ Wob,
                                float* __restrict__ out)  // [4096][768]
{
  const int lane = threadIdx.x & 31;
  const int wid  = threadIdx.x >> 5;
  const int hl   = lane >> 4;
  const int ln   = lane & 15;

  const int rowtile  = blockIdx.x / 3;
  const int colchunk = (blockIdx.x % 3) * 4 + wid;
  const int s0 = rowtile * 16;
  const int o0 = colchunk * 64;

  floatx8 acc[4];
#pragma unroll
  for (int t = 0; t < 4; ++t)
    acc[t] = (floatx8){0.f,0.f,0.f,0.f,0.f,0.f,0.f,0.f};

  const bf16* arow = attnb + (size_t)(s0 + ln) * DMODEL;
  const bf16* brow[4];
#pragma unroll
  for (int t = 0; t < 4; ++t)
    brow[t] = Wob + (size_t)(o0 + t * 16 + ln) * DMODEL;

#pragma unroll 2
  for (int kk = 0; kk < DMODEL; kk += 32) {
    bf16x16 a = ld16(arow + kk + hl * 8, arow + kk + 16 + hl * 8);
#pragma unroll
    for (int t = 0; t < 4; ++t) {
      bf16x16 b = ld16(brow[t] + kk + hl * 8, brow[t] + kk + 16 + hl * 8);
      acc[t] = wmma_bf16(a, b, acc[t]);
    }
  }

#pragma unroll
  for (int t = 0; t < 4; ++t)
#pragma unroll
    for (int r = 0; r < 8; ++r)
      out[(size_t)(s0 + r + 8 * hl) * DMODEL + (o0 + t * 16 + ln)] =
          acc[t][r];
}

// ---- host launch ---------------------------------------------------------

extern "C" void kernel_launch(void* const* d_in, const int* in_sizes, int n_in,
                              void* d_out, int out_size, void* d_ws,
                              size_t ws_size, hipStream_t stream) {
  const float* x  = (const float*)d_in[0];
  const float* Wq = (const float*)d_in[1];
  const float* Wk = (const float*)d_in[2];
  const float* Wv = (const float*)d_in[3];
  const float* Wo = (const float*)d_in[4];
  const int*   tp = (const int*)d_in[5];
  float* out = (float*)d_out;

  char* ws = (char*)d_ws;
  size_t off = 0;
  auto take = [&](size_t bytes) {
    size_t o = off; off += (bytes + 255) & ~(size_t)255; return o;
  };
  const size_t xElems = (size_t)SEQ * DMODEL;
  const size_t wElems = (size_t)DMODEL * DMODEL;
  const size_t hElems = (size_t)NHEAD * SEQ * DK;

  bf16* xb   = (bf16*)(ws + take(xElems * 2));
  bf16* Wqb  = (bf16*)(ws + take(wElems * 2));
  bf16* Wkb  = (bf16*)(ws + take(wElems * 2));
  bf16* Wvb  = (bf16*)(ws + take(wElems * 2));
  bf16* Wob  = (bf16*)(ws + take(wElems * 2));
  bf16* Qb   = (bf16*)(ws + take(hElems * 2));
  bf16* Kb   = (bf16*)(ws + take(hElems * 2));
  bf16* Vtb  = (bf16*)(ws + take(hElems * 2));
  bf16* attn = (bf16*)(ws + take(xElems * 2));

  // 0) fp32 -> bf16 conversions
  cvt_bf16<<<(int)((xElems + 255) / 256), 256, 0, stream>>>(x,  xb,  (int)xElems);
  cvt_bf16<<<(int)((wElems + 255) / 256), 256, 0, stream>>>(Wq, Wqb, (int)wElems);
  cvt_bf16<<<(int)((wElems + 255) / 256), 256, 0, stream>>>(Wk, Wkb, (int)wElems);
  cvt_bf16<<<(int)((wElems + 255) / 256), 256, 0, stream>>>(Wv, Wvb, (int)wElems);
  cvt_bf16<<<(int)((wElems + 255) / 256), 256, 0, stream>>>(Wo, Wob, (int)wElems);

  // 1) fused QKV projection + RoPE (Q scaled by 1/sqrt(dk))
  qkv_rope_kernel<<<3 * 256 * 3, 128, 0, stream>>>(xb, Wqb, Wkb, Wvb, tp,
                                                   Qb, Kb, Vtb);
  // 2) causal flash attention (async K/V staging into LDS)
  attention_kernel<<<NHEAD * 64, 128, 0, stream>>>(Qb, Kb, Vtb, attn);
  // 3) output projection
  out_proj_kernel<<<256 * 3, 128, 0, stream>>>(attn, Wob, out);
}
